// DiffusionNetBlock_66322884984851
// MI455X (gfx1250) — matrix operand order repaired and checked by reading, not compiled, not run on hardware
//
#include <hip/hip_runtime.h>
#include <hip/hip_bf16.h>

// DiffusionNet block for MI455X (gfx1250, wave32, WMMA bf16 path).
//  0) zero accumulation buffers (x_spec, gx/gy/gz)
//  1) prep: evecs -> bf16 (row-major evh + chunk-major evT), xm=mass*x_in ->
//     chunk-major xmc.  Chunk-major = [b][vchunk][feat][v%32]: reduce dim
//     innermost so WMMA fragments are two 16B runs per lane.
//  2) weight convert: A_weight / W0 / W1 -> bf16
//  3) x_spec[k,c] = sum_v evecs[v,k]*xm[v,c]   (split-K WMMA, pure global frags)
//  4) yT[c][k] = exp(-evals[k]*clamp(t[c])) * x_spec[k,c] -> bf16 (transposed)
//  5) x_diffuse[v,c] = sum_k evecs[v,k]*y[k,c] (WMMA; yT staged via TDM w/ pad)
//  6) gx/gy/gz = SpMM(grad*, x_diffuse)        (atomic scatter)
//  7) fused per-vertex: P = vflat@A^T (WMMA), gf=tanh(dot), f=[x|xd|gf],
//     h = relu(f@W0^T+b0) (WMMA), out = h@W1^T+b1+x_in (WMMA)

#define BB   4
#define VV   50000
#define VP   50016           // VV padded to a multiple of 32 (zero rows)
#define NCH  (VP / 32)       // 1563 v-chunks per batch
#define KK   128
#define CC   128
#define NNZE 800000
#define HIDN 256
#define MLPC 384
#define YLD  136             // padded yT row pitch (ushorts): 272B, 4-bank rot
#define FPAD 392             // padded feature row (ushorts): 784B
#define HPAD 264             // padded hidden row  (ushorts): 528B

typedef __attribute__((ext_vector_type(16))) __bf16       v16bf;
typedef __attribute__((ext_vector_type(8)))  float        v8f;
typedef __attribute__((ext_vector_type(4)))  unsigned int u32x4;
typedef __attribute__((ext_vector_type(8)))  int          i32x8;
typedef __attribute__((ext_vector_type(4)))  int          i32x4;

#if defined(__has_builtin)
#if __has_builtin(__builtin_amdgcn_tensor_load_to_lds) && \
    __has_builtin(__builtin_amdgcn_s_wait_tensorcnt)
#define HAVE_TDM 1
#endif
#endif
#ifndef HAVE_TDM
#define HAVE_TDM 0
#endif

union Frag { v16bf v; u32x4 q[2]; unsigned short s[16]; };

__device__ __forceinline__ unsigned short f2bfu(float f) {
  union { float f; unsigned u; } x; x.f = f;
  unsigned r = x.u + 0x7FFFu + ((x.u >> 16) & 1u);   // round-to-nearest-even
  return (unsigned short)(r >> 16);
}
__device__ __forceinline__ float bf2f(unsigned short h) {
  union { unsigned u; float f; } x; x.u = ((unsigned)h) << 16;
  return x.f;
}
__device__ __forceinline__ v8f wmma_bf16(v16bf a, v16bf b, v8f c) {
  return __builtin_amdgcn_wmma_f32_16x16x32_bf16(false, a, false, b, (short)0, c,
                                                 false, false);
}

// A-fragment (16x32 bf16), reduce dim contiguous in each row of `base`.
// lane<16: m=lane, elems [k0..7, k0+16..23]; lane>=16: [k0+8..15, k0+24..31]
template <typename T>
__device__ __forceinline__ v16bf fragA_row(const T* base, int ld, int lane) {
  Frag f;
  const T* p = base + (lane & 15) * ld + ((lane < 16) ? 0 : 8);
  f.q[0] = *(const u32x4*)(p);
  f.q[1] = *(const u32x4*)(p + 16);
  return f.v;
}
// B-fragment (32x16 bf16), reduce dim contiguous in each row of `base`
// (row n holds B[.][n]): lane<16 -> K [0..7, 16..23], lane>=16 -> K [8..15, 24..31].
template <typename T>
__device__ __forceinline__ v16bf fragB_row(const T* base, int ld, int lane) {
  Frag f;
  const T* p = base + (lane & 15) * ld + ((lane < 16) ? 0 : 8);
  f.q[0] = *(const u32x4*)(p);
  f.q[1] = *(const u32x4*)(p + 16);
  return f.v;
}

// ---------------------------------------------------------------- kernels ---

__global__ void k_zero(u32x4* p, int n4) {
  u32x4 z = {0u, 0u, 0u, 0u};
  for (int i = blockIdx.x * blockDim.x + threadIdx.x; i < n4;
       i += gridDim.x * blockDim.x)
    p[i] = z;
}

// one block per padded (b*VP + v); 128 threads over channels
__global__ void k_prep(const float* __restrict__ x_in,
                       const float* __restrict__ evecs,
                       const float* __restrict__ mass,
                       unsigned short* __restrict__ evh,   // [b][v][k] row-major
                       unsigned short* __restrict__ evT,   // [b][ch][k][32]
                       unsigned short* __restrict__ xmc) { // [b][ch][c][32]
  size_t g = blockIdx.x;                 // 0 .. B*VP-1
  int b = (int)(g / VP);
  int v = (int)(g - (size_t)b * VP);
  int t = threadIdx.x;
  int ch = v >> 5, r = v & 31;
  unsigned short ev = 0, xm = 0;
  if (v < VV) {
    size_t s = ((size_t)b * VV + v);
    ev = f2bfu(evecs[s * KK + t]);
    xm = f2bfu(mass[s] * x_in[s * CC + t]);
  }
  evh[g * KK + t] = ev;
  evT[((size_t)(b * NCH + ch) * KK + t) * 32 + r] = ev;
  xmc[((size_t)(b * NCH + ch) * CC + t) * 32 + r] = xm;
}

__global__ void k_cvt(const float* __restrict__ a, const float* __restrict__ w0,
                      const float* __restrict__ w1,
                      unsigned short* __restrict__ ah,
                      unsigned short* __restrict__ w0h,
                      unsigned short* __restrict__ w1h) {
  int i = blockIdx.x * blockDim.x + threadIdx.x;
  if (i < MLPC * MLPC) ah[i]  = f2bfu(a[i]);
  if (i < HIDN * MLPC) w0h[i] = f2bfu(w0[i]);
  if (i < CC * HIDN)   w1h[i] = f2bfu(w1[i]);
}

// x_spec: split-K over V, fragments straight from chunk-major global arrays.
// grid = (8 k-tiles, B, SPLIT), block = 256 (8 waves = 8 c-tiles). No LDS.
#define XS_SPLIT 32
__global__ void k_xspec(const unsigned short* __restrict__ evT,
                        const unsigned short* __restrict__ xmc,
                        float* __restrict__ xspec) {
  int kt = blockIdx.x, b = blockIdx.y, sp = blockIdx.z;
  int tid = threadIdx.x, lane = tid & 31, w = tid >> 5;
  v8f acc = {0.f, 0.f, 0.f, 0.f, 0.f, 0.f, 0.f, 0.f};
  for (int ch = sp; ch < NCH; ch += XS_SPLIT) {
    const unsigned short* Ab = evT + ((size_t)(b * NCH + ch) * KK + kt * 16) * 32;
    const unsigned short* Bb = xmc + ((size_t)(b * NCH + ch) * CC + w * 16) * 32;
    __builtin_prefetch(Ab + (size_t)XS_SPLIT * KK * 32, 0, 1);
    __builtin_prefetch(Bb + (size_t)XS_SPLIT * CC * 32, 0, 1);
    v16bf a  = fragA_row(Ab, 32, lane);  // A[m=k][kk=v]
    v16bf bf = fragB_row(Bb, 32, lane);  // B[kk=v][n=c]
    acc = wmma_bf16(a, bf, acc);
  }
  int ccol = w * 16 + (lane & 15);
#pragma unroll
  for (int i = 0; i < 8; ++i) {
    int krow = kt * 16 + i + ((lane >= 16) ? 8 : 0);
    atomicAdd(&xspec[((size_t)b * KK + krow) * CC + ccol], acc[i]);
  }
}

// yT[b][c][k] = exp(-evals[k]*clamp(t[c])) * x_spec[b][k][c]  (transposed out)
__global__ void k_coef(const float* __restrict__ evals,
                       const float* __restrict__ tdiff,
                       const float* __restrict__ xspec,
                       unsigned short* __restrict__ ybfT) {
  int i = blockIdx.x * 256 + threadIdx.x;  // B*K*C = 65536
  int c = i & 127, k = (i >> 7) & 127, b = i >> 14;
  float t = fmaxf(tdiff[c], 1e-8f);
  float coef = expf(-evals[b * KK + k] * t);
  ybfT[((size_t)b * CC + c) * KK + k] = f2bfu(coef * xspec[i]);
}

// x_diffuse: grid = (ceil(3125/8), B), block 256; wave w owns m-tile (blk*8+w)
__global__ void k_fromb(const unsigned short* __restrict__ evh,
                        const unsigned short* __restrict__ ybfT,
                        float* __restrict__ xdif) {
  __shared__ alignas(16) unsigned short Ybuf[CC * YLD];  // yT tile, padded rows
  int b = blockIdx.y;
  int tid = threadIdx.x, lane = tid & 31, w = tid >> 5;
  const unsigned short* src = ybfT + (size_t)b * CC * KK;
#if HAVE_TDM
  // TDM: DMA the 128x64-dword yT tile into LDS with 4-dword padding every
  // 64 dwords (row pitch 272B -> 4-bank rotation, conflict-free b128 reads).
  // Issued by wave 0 only (TDM ignores EXEC; one descriptor per workgroup).
  if (w == 0) {
    unsigned lds_addr = (unsigned)(size_t)(void*)&Ybuf[0];
    unsigned long long ga = (unsigned long long)(size_t)(const void*)src;
    // group0: count=1 | lds_addr | global_addr[56:0] | type=2
    u32x4 g0 = {1u, lds_addr, (unsigned)ga,
                (unsigned)((ga >> 32) & 0x01FFFFFFu) | (2u << 30)};
    // group1: data_size=4B | pad_enable | pad_interval=5 (64 dw) |
    //         pad_amount=3 (4 dw); tensor 64x128 dwords; tile = whole tensor
    int d0 = (int)((2u << 16) | (1u << 20) | (5u << 22) | (3u << 25));
    i32x8 g1 = {d0, (int)(64u << 16), (int)(128u << 16),
                (int)(64u << 16), 128, 64, 0, 0};
    i32x4 g2 = {0, 0, 0, 0};
    i32x4 g3 = {0, 0, 0, 0};
    i32x8 g4 = {0, 0, 0, 0, 0, 0, 0, 0};
    __builtin_amdgcn_tensor_load_to_lds(g0, g1, g2, g3, g4, 0);
    __builtin_amdgcn_s_wait_tensorcnt((short)0);
  }
#else
  for (int i = tid; i < CC * 16; i += 256) {    // 2048 b128 moves, padded rows
    int r = i >> 4, h = i & 15;
    *(u32x4*)(&Ybuf[r * YLD + h * 8]) = *(const u32x4*)(src + r * KK + h * 8);
  }
#endif
  __syncthreads();
  int mt = blockIdx.x * 8 + w;
  bool active = (mt < VV / 16);  // wave-uniform => EXEC stays all-ones inside
  const unsigned short* Erow = evh + ((size_t)b * VP + (size_t)mt * 16) * KK;
  v16bf afrag[4];
  if (active) {
#pragma unroll
    for (int kk = 0; kk < 4; ++kk)
      afrag[kk] = fragA_row(Erow + kk * 32, KK, lane);
  }
  for (int ct = 0; ct < 8; ++ct) {
    v8f acc = {0.f, 0.f, 0.f, 0.f, 0.f, 0.f, 0.f, 0.f};
    if (active) {
#pragma unroll
      for (int kk = 0; kk < 4; ++kk) {
        // B[kk][n] = y[kc+kk][c0+n] = yT[c0+n][kc+kk]: contiguous LDS rows
        v16bf bf = fragB_row(&Ybuf[(size_t)(ct * 16) * YLD + kk * 32], YLD, lane);
        acc = wmma_bf16(afrag[kk], bf, acc);
      }
      int col = ct * 16 + (lane & 15);
#pragma unroll
      for (int i = 0; i < 8; ++i) {
        int row = i + ((lane >= 16) ? 8 : 0);
        xdif[((size_t)b * VV + (size_t)mt * 16 + row) * CC + col] = acc[i];
      }
    }
  }
}

// SpMM scatter: grid = (NNZ/8, B, 3), block 128 (one channel per thread)
__global__ void k_spmm(const float* vX, const int* rX, const int* cX,
                       const float* vY, const int* rY, const int* cY,
                       const float* vZ, const int* rZ, const int* cZ,
                       const float* __restrict__ xdif,
                       float* gx, float* gy, float* gz) {
  int dir = blockIdx.z, b = blockIdx.y, c = threadIdx.x;
  const float* vals = (dir == 0) ? vX : (dir == 1) ? vY : vZ;
  const int*   rows = (dir == 0) ? rX : (dir == 1) ? rY : rZ;
  const int*   cols = (dir == 0) ? cX : (dir == 1) ? cY : cZ;
  float* g = (dir == 0) ? gx : (dir == 1) ? gy : gz;
  size_t eb = (size_t)b * NNZE + (size_t)blockIdx.x * 8;
  size_t vb = (size_t)b * VV;
#pragma unroll
  for (int e = 0; e < 8; ++e) {
    float val = vals[eb + e];
    int r = rows[eb + e], co = cols[eb + e];
    if (e + 1 < 8) __builtin_prefetch(&xdif[(vb + cols[eb + e + 1]) * CC + c], 0, 1);
    atomicAdd(&g[(vb + r) * CC + c], val * xdif[(vb + co) * CC + c]);
  }
}

// Fused per-vertex tail: grid = B*V/16 blocks of 256 (8 waves), 16 vertices/block
__global__ void k_mlp(const float* __restrict__ x_in,
                      const float* __restrict__ xdif,
                      const float* __restrict__ gx, const float* __restrict__ gy,
                      const float* __restrict__ gz,
                      const unsigned short* __restrict__ Ah,
                      const unsigned short* __restrict__ W0h,
                      const unsigned short* __restrict__ W1h,
                      const float* __restrict__ b0, const float* __restrict__ b1,
                      float* __restrict__ out) {
  __shared__ alignas(16) unsigned short vf[16][FPAD];  // vflat (3c+d), bf16
  __shared__ alignas(16) unsigned short fb[16][FPAD];  // [x_in|x_diffuse|gf]
  __shared__ alignas(16) float          pb[16][MLPC];  // vflat @ A^T, f32
  __shared__ alignas(16) unsigned short hb[16][HPAD];  // hidden, bf16
  int tid = threadIdx.x, lane = tid & 31, w = tid >> 5;
  size_t g0 = (size_t)blockIdx.x * 16;

  for (int idx = tid; idx < 16 * CC; idx += 256) {
    int m = idx >> 7, c = idx & 127;
    size_t gi = (g0 + m) * CC + c;
    fb[m][c]         = f2bfu(x_in[gi]);
    fb[m][CC + c]    = f2bfu(xdif[gi]);
    vf[m][3 * c + 0] = f2bfu(gx[gi]);
    vf[m][3 * c + 1] = f2bfu(gy[gi]);
    vf[m][3 * c + 2] = f2bfu(gz[gi]);
  }
  __syncthreads();

  // GEMM1: pb = vf @ A^T   (16 x 384)
  for (int nt = w; nt < MLPC / 16; nt += 8) {
    v8f acc = {0.f, 0.f, 0.f, 0.f, 0.f, 0.f, 0.f, 0.f};
#pragma unroll
    for (int kk = 0; kk < MLPC / 32; ++kk) {
      v16bf a  = fragA_row(&vf[0][0] + kk * 32, FPAD, lane);
      v16bf bf = fragB_row(Ah + (size_t)(nt * 16) * MLPC + kk * 32, MLPC, lane);
      acc = wmma_bf16(a, bf, acc);
    }
    int col = nt * 16 + (lane & 15);
#pragma unroll
    for (int i = 0; i < 8; ++i) pb[i + ((lane >= 16) ? 8 : 0)][col] = acc[i];
  }
  __syncthreads();

  // grad features: gf = tanh(sum_d vflat[3c+d]*P[3c+d])
  for (int idx = tid; idx < 16 * CC; idx += 256) {
    int m = idx >> 7, c = idx & 127;
    float s = bf2f(vf[m][3 * c + 0]) * pb[m][3 * c + 0] +
              bf2f(vf[m][3 * c + 1]) * pb[m][3 * c + 1] +
              bf2f(vf[m][3 * c + 2]) * pb[m][3 * c + 2];
    fb[m][2 * CC + c] = f2bfu(tanhf(s));
  }
  __syncthreads();

  // GEMM2: hb = relu(fb @ W0^T + b0)   (16 x 256)
  for (int nt = w; nt < HIDN / 16; nt += 8) {
    v8f acc = {0.f, 0.f, 0.f, 0.f, 0.f, 0.f, 0.f, 0.f};
#pragma unroll
    for (int kk = 0; kk < MLPC / 32; ++kk) {
      v16bf a  = fragA_row(&fb[0][0] + kk * 32, FPAD, lane);
      v16bf bf = fragB_row(W0h + (size_t)(nt * 16) * MLPC + kk * 32, MLPC, lane);
      acc = wmma_bf16(a, bf, acc);
    }
    int col = nt * 16 + (lane & 15);
    float bias = b0[col];
#pragma unroll
    for (int i = 0; i < 8; ++i)
      hb[i + ((lane >= 16) ? 8 : 0)][col] = f2bfu(fmaxf(acc[i] + bias, 0.f));
  }
  __syncthreads();

  // GEMM3: out = hb @ W1^T + b1 + x_in   (16 x 128), one n-tile per wave
  {
    int nt = w;
    v8f acc = {0.f, 0.f, 0.f, 0.f, 0.f, 0.f, 0.f, 0.f};
#pragma unroll
    for (int kk = 0; kk < HIDN / 32; ++kk) {
      v16bf a  = fragA_row(&hb[0][0] + kk * 32, HPAD, lane);
      v16bf bf = fragB_row(W1h + (size_t)(nt * 16) * HIDN + kk * 32, HIDN, lane);
      acc = wmma_bf16(a, bf, acc);
    }
    int col = nt * 16 + (lane & 15);
    float bias = b1[col];
#pragma unroll
    for (int i = 0; i < 8; ++i) {
      int row = i + ((lane >= 16) ? 8 : 0);
      size_t gi = (g0 + row) * CC + col;
      out[gi] = acc[i] + bias + x_in[gi];
    }
  }
}

// ------------------------------------------------------------------ launch ---

extern "C" void kernel_launch(void* const* d_in, const int* in_sizes, int n_in,
                              void* d_out, int out_size, void* d_ws,
                              size_t ws_size, hipStream_t stream) {
  const float* x_in  = (const float*)d_in[0];
  const float* evals = (const float*)d_in[1];
  const float* evecs = (const float*)d_in[2];
  const float* mass  = (const float*)d_in[3];
  const float* gXv = (const float*)d_in[4];  const int* gXr = (const int*)d_in[5];  const int* gXc = (const int*)d_in[6];
  const float* gYv = (const float*)d_in[7];  const int* gYr = (const int*)d_in[8];  const int* gYc = (const int*)d_in[9];
  const float* gZv = (const float*)d_in[10]; const int* gZr = (const int*)d_in[11]; const int* gZc = (const int*)d_in[12];
  const float* tdif = (const float*)d_in[13];
  const float* Aw = (const float*)d_in[14];
  const float* W0 = (const float*)d_in[15];
  const float* b0 = (const float*)d_in[16];
  const float* W1 = (const float*)d_in[17];
  const float* b1 = (const float*)d_in[18];
  float* out = (float*)d_out;

  char* ws = (char*)d_ws;
  size_t off = 0;
  auto carve = [&](size_t bytes) -> void* {
    void* p = ws + off;
    off += (bytes + 255) & ~(size_t)255;
    return p;
  };
  // atomically-accumulated buffers first and contiguous (single memset)
  float* gx    = (float*)carve(sizeof(float) * (size_t)BB * VV * CC);
  float* gy    = (float*)carve(sizeof(float) * (size_t)BB * VV * CC);
  float* gz    = (float*)carve(sizeof(float) * (size_t)BB * VV * CC);
  float* xspec = (float*)carve(sizeof(float) * (size_t)BB * KK * CC);
  size_t zero_bytes = off;
  float* xdif  = (float*)carve(sizeof(float) * (size_t)BB * VV * CC);
  unsigned short* evh  = (unsigned short*)carve(2ull * BB * VP * KK);
  unsigned short* evT  = (unsigned short*)carve(2ull * BB * VP * KK);
  unsigned short* xmc  = (unsigned short*)carve(2ull * BB * VP * CC);
  unsigned short* ybfT = (unsigned short*)carve(2ull * BB * KK * CC);
  unsigned short* Ah   = (unsigned short*)carve(2ull * MLPC * MLPC);
  unsigned short* W0h  = (unsigned short*)carve(2ull * HIDN * MLPC);
  unsigned short* W1h  = (unsigned short*)carve(2ull * CC * HIDN);

  int n4 = (int)(zero_bytes / 16);
  k_zero<<<4096, 256, 0, stream>>>((u32x4*)ws, n4);

  k_prep<<<BB * VP, CC, 0, stream>>>(x_in, evecs, mass, evh, evT, xmc);

  k_cvt<<<(MLPC * MLPC + 255) / 256, 256, 0, stream>>>(Aw, W0, W1, Ah, W0h, W1h);

  k_xspec<<<dim3(KK / 16, BB, XS_SPLIT), 256, 0, stream>>>(evT, xmc, xspec);

  k_coef<<<(BB * KK * CC) / 256, 256, 0, stream>>>(evals, tdif, xspec, ybfT);

  k_fromb<<<dim3((VV / 16 + 7) / 8, BB), 256, 0, stream>>>(evh, ybfT, xdif);

  k_spmm<<<dim3(NNZE / 8, BB, 3), CC, 0, stream>>>(gXv, gXr, gXc, gYv, gYr, gYc,
                                                   gZv, gZr, gZc, xdif, gx, gy, gz);

  k_mlp<<<(BB * VV) / 16, 256, 0, stream>>>(x_in, xdif, gx, gy, gz, Ah, W0h, W1h,
                                            b0, b1, out);
}
